// EquivariantAttention_10342281249302
// MI455X (gfx1250) — matrix operand order repaired and checked by reading, not compiled, not run on hardware
//
#include <hip/hip_runtime.h>
#include <math.h>

typedef __attribute__((ext_vector_type(16))) __bf16 v16bf;
typedef __attribute__((ext_vector_type(8)))  float  v8f;

#define KNBR 16
#define W2_CHUNKS 33                    // 32 real K-chunks of 32 + 1 bias-fold chunk
#define W2IMG_ELEMS (W2_CHUNKS*3*32*16) // per-lane register-image, 16 bf16 per lane
#define W1IMG_ELEMS (4*32*16)

// Branch-free erf (Abramowitz-Stegun 7.1.26, max abs err ~1.5e-7).
// Uses hw v_rcp_f32 / v_exp_f32 (TRANS ops co-execute with XDL WMMA).
__device__ __forceinline__ float erf_fast(float x) {
    float ax = fabsf(x);
    float t  = __builtin_amdgcn_rcpf(fmaf(0.3275911f, ax, 1.0f));
    float p  = fmaf(fmaf(fmaf(fmaf(1.061405429f, t, -1.453152027f), t,
                               1.421413741f), t, -0.284496736f), t, 0.254829592f);
    float er = 1.0f - (p * t) * __expf(-ax * ax);
    return copysignf(er, x);
}
__device__ __forceinline__ float gelu_exact(float x) {
    return 0.5f * x * (1.0f + erf_fast(x * 0.7071067811865475f));
}

// ---------------------------------------------------------------------------
// Prep: pack W1 and W2' (with b2_lin folded in as K-chunk 32) into the exact
// per-lane WMMA B-operand register image (gfx1250 layout: lanes 0-15 hold
// K {0..7,16..23}, lanes 16-31 hold K {8..15,24..31}; slot s<8 -> K 8*half+s, s>=8 -> K 8+8*half+s).
// ---------------------------------------------------------------------------
__global__ void eqattn_prep(const float* __restrict__ W1,
                            const float* __restrict__ W2,
                            const float* __restrict__ b2_lin,
                            unsigned short* __restrict__ w2img,
                            unsigned short* __restrict__ w1img)
{
    int tid    = blockIdx.x * blockDim.x + threadIdx.x;
    int stride = gridDim.x * blockDim.x;

    // w2img[i], i = ((t*3 + ot)*32 + lane)*16 + s
    for (int i = tid; i < W2IMG_ELEMS; i += stride) {
        int s  = i & 15;
        int L  = (i >> 4) & 31;
        int ot = (i >> 9) % 3;
        int t  = i / (3 * 32 * 16);
        int o  = ot * 16 + (L & 15);     // output channel 0..47 (N index)
        int c  = 2 * t + (s >> 3);       // "h channel": 0..63 real, 64 = bias row
        int jj = (L >> 4) * 8 + (s & 7);
        float v;
        if (c < 64)       v = W2[(size_t)(o * 16 + jj) * 64 + c]; // W2[(o,j=jj), c]
        else if (c == 64) v = b2_lin[o * 16 + jj];
        else              v = 0.0f;
        __bf16 b = (__bf16)v;
        w2img[i] = __builtin_bit_cast(unsigned short, b);
    }

    // w1img[i], i = (ot*32 + lane)*16 + s ; B operand for layer-1 (K=32, N=16)
    for (int i = tid; i < W1IMG_ELEMS; i += stride) {
        int s  = i & 15;
        int L  = (i >> 4) & 31;
        int ot = i >> 9;
        int cN = ot * 16 + (L & 15);               // hidden channel 0..63
        int Kk = ((s >> 3) ? 16 : 0) + (L >> 4) * 8 + (s & 7); // edge-feature index 0..31
        __bf16 b = (__bf16)W1[cN * 32 + Kk];
        w1img[i] = __builtin_bit_cast(unsigned short, b);
    }
}

// ---------------------------------------------------------------------------
// Main fused kernel: one wave32 per node. 16 WMMA rows == node's 16 neighbors.
// ---------------------------------------------------------------------------
__global__ __launch_bounds__(32) void eqattn_main(
    const float* __restrict__ b1,        // (N,16,4,2)
    const float* __restrict__ b2,        // (N,16,2,4)
    const float* __restrict__ ef,        // (N,16,32)
    const float* __restrict__ f,         // (N,8,4)
    const int*   __restrict__ nbr,       // (N,16)
    const float* __restrict__ b1_lin,    // (64)
    const float* __restrict__ W_out,     // (16,8)
    const float* __restrict__ bias_out,  // (8,1)
    const unsigned short* __restrict__ w2img_u,
    const unsigned short* __restrict__ w1img_u,
    float* __restrict__ out)             // (N,8,4)
{
    const int n  = blockIdx.x;
    const int L  = threadIdx.x;
    const int e  = L & 15;   // edge row (WMMA M) owned by this lane
    const int hi = L >> 4;   // which K-half / M-half this lane carries

    __shared__ float hlds[16 * 66];      // h (f32) + columns 64(=1),65(=0) for bias fold
    __shared__ float t2[16 * 48];        // tmp2
    __shared__ float qlds[16 * 24 * 4];  // qkv
    __shared__ float b2lds[16 * 8];      // b2f tile
    __shared__ float sclds[16 * 4];      // attention scores
    __shared__ float outlds[32];         // attention output (m,d)

    const v16bf* w2p = (const v16bf*)w2img_u;
    const v16bf* w1p = (const v16bf*)w1img_u;

    // ---- Layer 1: h = gelu(ef @ W1^T + b1_lin), 4 WMMAs (K=32, N=16 each) ----
    // A operand (16-bit A layout): lane<16 carries K {0..7,16..23}, lane>=16 {8..15,24..31}
    const float* efrow = ef + (size_t)(n * KNBR + e) * 32;
    const int kbase = hi * 8;
    v16bf a1;
#pragma unroll
    for (int s = 0; s < 8; ++s) a1[s]     = (__bf16)efrow[kbase + s];
#pragma unroll
    for (int s = 0; s < 8; ++s) a1[8 + s] = (__bf16)efrow[kbase + 16 + s];

#pragma unroll
    for (int o4 = 0; o4 < 4; ++o4) {
        float bv = b1_lin[o4 * 16 + e];          // bias of this lane's N column
        v8f c = {bv, bv, bv, bv, bv, bv, bv, bv};
        v16bf bb = w1p[o4 * 32 + L];
        c = __builtin_amdgcn_wmma_f32_16x16x32_bf16(false, a1, false, bb,
                                                    (short)0, c, false, false);
#pragma unroll
        for (int r = 0; r < 8; ++r) {            // C layout: row = r + 8*hi, col = o4*16 + e
            hlds[(r + 8 * hi) * 66 + o4 * 16 + e] = gelu_exact(c[r]);
        }
    }
    if (hi == 0) { hlds[e * 66 + 64] = 1.0f; hlds[e * 66 + 65] = 0.0f; }

    // ---- tmp[e,j] = sum_d f_src[e,m,d] * b1f[e,d,l],  j = m*2+l (this lane's half) ----
    const int eg = n * KNBR + e;
    const int src = nbr[eg];
    const float* b1e = b1 + (size_t)eg * 8;
    const float* fs  = f  + (size_t)src * 32;
    float tmpv[8];
#pragma unroll
    for (int jj = 0; jj < 8; ++jj) {
        int j = hi * 8 + jj, m = j >> 1, l = j & 1;
        float s = 0.0f;
#pragma unroll
        for (int d = 0; d < 4; ++d) s += fs[m * 4 + d] * b1e[d * 2 + l];
        tmpv[jj] = s;
    }
    __syncthreads();   // hlds visible across lanes (single-wave WG: barrier == s_nop + waits)

    // ---- Big fused GEMM: tmp2[16,48] = X[16,1056] @ W2'[1056,48]
    //      X[e, c*16+j] = h[e,c]*tmp[e,j]; slot s -> (c = 2t + s/8, j = lane-half + s%8)
    v8f acc0 = {0,0,0,0,0,0,0,0}, acc1 = {0,0,0,0,0,0,0,0}, acc2 = {0,0,0,0,0,0,0,0};
    for (int t = 0; t < W2_CHUNKS; ++t) {
        if (t + 1 < W2_CHUNKS)
            __builtin_prefetch(&w2p[((t + 1) * 3) * 32 + L], 0, 3);
        float h0 = hlds[e * 66 + 2 * t];
        float h1 = hlds[e * 66 + 2 * t + 1];
        v16bf x;
#pragma unroll
        for (int s = 0; s < 8; ++s) x[s]     = (__bf16)(h0 * tmpv[s]);
#pragma unroll
        for (int s = 0; s < 8; ++s) x[8 + s] = (__bf16)(h1 * tmpv[s]);
        v16bf bb0 = w2p[(t * 3 + 0) * 32 + L];
        v16bf bb1 = w2p[(t * 3 + 1) * 32 + L];
        v16bf bb2 = w2p[(t * 3 + 2) * 32 + L];
        acc0 = __builtin_amdgcn_wmma_f32_16x16x32_bf16(false, x, false, bb0, (short)0, acc0, false, false);
        acc1 = __builtin_amdgcn_wmma_f32_16x16x32_bf16(false, x, false, bb1, (short)0, acc1, false, false);
        acc2 = __builtin_amdgcn_wmma_f32_16x16x32_bf16(false, x, false, bb2, (short)0, acc2, false, false);
    }

    // spill tmp2 to LDS (C layout: row r+8*hi, col = ot*16 + e)
#pragma unroll
    for (int r = 0; r < 8; ++r) {
        int er = r + 8 * hi;
        t2[er * 48 +  0 + e] = acc0[r];
        t2[er * 48 + 16 + e] = acc1[r];
        t2[er * 48 + 32 + e] = acc2[r];
    }
    // stage b2f tile (16 edges x 2 x 4)
    for (int i = L; i < 128; i += 32) b2lds[i] = b2[(size_t)n * 128 + i];
    __syncthreads();

    // ---- qkv[e,oo,d] = t2[e,2oo]*b2f[e,0,d] + t2[e,2oo+1]*b2f[e,1,d] ----
    for (int idx = L; idx < 16 * 24 * 4; idx += 32) {
        int ee = idx / 96, rem = idx % 96, oo = rem >> 2, d = rem & 3;
        float v = t2[ee * 48 + oo * 2] * b2lds[ee * 8 + d]
                + t2[ee * 48 + oo * 2 + 1] * b2lds[ee * 8 + 4 + d];
        qlds[(ee * 24 + oo) * 4 + d] = v;
    }
    __syncthreads();

    // ---- scores[e,h] = scale * <q[e,h,:], k[e,h,:]> ----
    for (int p = L; p < 64; p += 32) {
        int ee = p >> 2, hh = p & 3;
        float s = 0.0f;
#pragma unroll
        for (int hd = 0; hd < 8; ++hd) {
            int m = hh * 2 + (hd >> 2), d = hd & 3;
            s += qlds[(ee * 24 + m) * 4 + d] * qlds[(ee * 24 + 8 + m) * 4 + d];
        }
        sclds[ee * 4 + hh] = s * 0.3535533905932738f;   // 8^-0.5
    }
    __syncthreads();

    // ---- softmax over the 16 neighbors + weighted sum of v ----
    {
        int hh = L >> 3, hd = L & 7;
        int m = hh * 2 + (hd >> 2), d = hd & 3;
        float mx = -1e30f;
#pragma unroll
        for (int ee = 0; ee < 16; ++ee) mx = fmaxf(mx, sclds[ee * 4 + hh]);
        float den = 0.0f, ov = 0.0f;
#pragma unroll
        for (int ee = 0; ee < 16; ++ee) {
            float w = __expf(sclds[ee * 4 + hh] - mx);   // hw v_exp_f32
            den += w;
            ov  += w * qlds[(ee * 24 + 16 + m) * 4 + d];
        }
        outlds[m * 4 + d] = ov / den;
    }
    __syncthreads();

    // ---- out_proj + irrep select (INDICES = [0,1,1,1]) + scalar bias ----
    {
        int m = L >> 2, d = L & 3;
        int l = (d == 0) ? 0 : 1;
        float s = 0.0f;
#pragma unroll
        for (int mp = 0; mp < 8; ++mp)
            s += W_out[(l * 8 + m) * 8 + mp] * outlds[mp * 4 + d];
        if (d == 0) s += bias_out[m];
        out[(size_t)n * 32 + L] = s;
    }
}

extern "C" void kernel_launch(void* const* d_in, const int* in_sizes, int n_in,
                              void* d_out, int out_size, void* d_ws, size_t ws_size,
                              hipStream_t stream) {
    (void)n_in; (void)out_size; (void)ws_size;
    const float* b1       = (const float*)d_in[0];
    const float* b2       = (const float*)d_in[1];
    const float* ef       = (const float*)d_in[2];
    const float* f        = (const float*)d_in[3];
    const int*   nbr      = (const int*)  d_in[4];
    const float* W1       = (const float*)d_in[5];
    const float* b1_lin   = (const float*)d_in[6];
    const float* W2       = (const float*)d_in[7];
    const float* b2_lin   = (const float*)d_in[8];
    const float* W_out    = (const float*)d_in[9];
    const float* bias_out = (const float*)d_in[10];

    unsigned short* w2img = (unsigned short*)d_ws;
    unsigned short* w1img = w2img + W2IMG_ELEMS;

    const int N = in_sizes[4] / KNBR;   // neighbor_idx has N*K elements

    eqattn_prep<<<64, 256, 0, stream>>>(W1, W2, b2_lin, w2img, w1img);
    eqattn_main<<<N, 32, 0, stream>>>(b1, b2, ef, f, nbr, b1_lin, W_out, bias_out,
                                      w2img, w1img, (float*)d_out);
}
